// CausalSelfAttention_30880814858984
// MI455X (gfx1250) — matrix operand order repaired and checked by reading, not compiled, not run on hardware
//
#include <hip/hip_runtime.h>
#include <hip/hip_bf16.h>

typedef __bf16 bf16;
typedef __attribute__((ext_vector_type(16))) __bf16 v16bf;
typedef __attribute__((ext_vector_type(8)))  float  v8f;

constexpr int BB = 4;      // batch
constexpr int TT = 2048;   // seq len
constexpr int CC = 1024;   // channels
constexpr int HH = 16;     // heads
constexpr int DD = 64;     // head dim

// ---------------------------------------------------------------------------
// WMMA helpers (CDNA5 gfx1250, wave32, 16x16x32 bf16 -> f32)
// ---------------------------------------------------------------------------
__device__ __forceinline__ v8f wmma_bf16(v16bf a, v16bf b, v8f c) {
  // 8 args: (neg_a, A, neg_b, B, c_mod, C, reuse_a, reuse_b)
  return __builtin_amdgcn_wmma_f32_16x16x32_bf16(false, a, false, b,
                                                 (short)0, c, false, false);
}

// A fragment: 16x32 bf16 tile, A row-major with leading dim lda, tile row 0 at A.
// K order used here (contiguous, same as load_bt_frag): lanes 0-15 hold M=lane, K = 0..15; lanes 16-31: K = 16..31.
__device__ __forceinline__ v16bf load_a_frag(const bf16* __restrict__ A, int lda, int k0) {
  const int lane = threadIdx.x & 31;
  const int m    = lane & 15;
  const int kb   = (lane & 16) ? 16 : 0;
  const bf16* p  = A + (size_t)m * lda + k0 + kb;
  v16bf f;
#pragma unroll
  for (int i = 0; i < 8; ++i) f[i] = p[i];
#pragma unroll
  for (int i = 0; i < 8; ++i) f[8 + i] = p[8 + i];
  return f;
}

// B fragment: 32x16 bf16, sourced from BT stored N-major (BT[n*ldb + k]).
// K order used here (contiguous, same as load_a_frag): lane col = lane%16; lanes 0-15 hold K=0..15, lanes 16-31 K=16..31.
__device__ __forceinline__ v16bf load_bt_frag(const bf16* __restrict__ BT, int ldb, int k0) {
  const int lane = threadIdx.x & 31;
  const int n    = lane & 15;
  const int kb   = (lane & 16) ? 16 : 0;
  const bf16* p  = BT + (size_t)n * ldb + k0 + kb;
  v16bf f;
#pragma unroll
  for (int i = 0; i < 16; ++i) f[i] = p[i];
  return f;
}

// ---------------------------------------------------------------------------
// fp32 -> bf16 conversion / transposing conversion
// ---------------------------------------------------------------------------
__global__ void cvt_bf16_kernel(const float* __restrict__ in, bf16* __restrict__ out, long n) {
  long i = (long)blockIdx.x * blockDim.x + threadIdx.x;
  long stride = (long)gridDim.x * blockDim.x;
  for (; i < n; i += stride) out[i] = (bf16)in[i];
}

// out[c * rows + r] = in[r * cols + c]   (weights -> N-major bf16)
__global__ void transpose_cvt_kernel(const float* __restrict__ in, bf16* __restrict__ out,
                                     int rows, int cols) {
  long n = (long)rows * cols;
  long i = (long)blockIdx.x * blockDim.x + threadIdx.x;
  long stride = (long)gridDim.x * blockDim.x;
  for (; i < n; i += stride) {
    int c = (int)(i % cols);
    int r = (int)(i / cols);
    out[(size_t)c * rows + r] = (bf16)in[i];
  }
}

// ---------------------------------------------------------------------------
// QKV GEMM: [8192 x 1024] @ [1024 x 3072] + bias, scatter to q/k/vt bf16 bufs.
// block = 256 threads = 8 waves arranged 2(M) x 4(N); wave tile = 16 x 64
// (one A-fragment reused across 4 B-fragments -> 4 WMMAs per K-step).
// block tile = 32 x 256.
// ---------------------------------------------------------------------------
__global__ void __launch_bounds__(256) qkv_gemm_kernel(
    const bf16* __restrict__ Xb,    // [B*T, C]
    const bf16* __restrict__ WT,    // [3C, C]  (N-major)
    const float* __restrict__ bias, // [3C]
    bf16* __restrict__ qbuf,        // [B,H,T,D]  (pre-scaled by 1/sqrt(D))
    bf16* __restrict__ kbuf,        // [B,H,T,D]
    bf16* __restrict__ vtbuf)       // [B,H,D,T]
{
  const int lane = threadIdx.x & 31;
  const int wave = threadIdx.x >> 5;
  const int m0 = blockIdx.x * 32 + (wave >> 2) * 16;
  const int n0 = blockIdx.y * 256 + (wave & 3) * 64;

  const bf16* Arow = Xb + (size_t)m0 * CC;
  const bf16* Brow = WT + (size_t)n0 * CC;

  v8f acc[4];
#pragma unroll
  for (int j = 0; j < 4; ++j) acc[j] = (v8f){};

  for (int k0 = 0; k0 < CC; k0 += 32) {
    const v16bf a = load_a_frag(Arow, CC, k0);
#pragma unroll
    for (int j = 0; j < 4; ++j) {
      v16bf b = load_bt_frag(Brow + (size_t)(j * 16) * CC, CC, k0);
      acc[j] = wmma_bf16(a, b, acc[j]);
    }
  }

  // Epilogue: 64-wide wave tile spans exactly one head and one of q/k/v.
  const int half = (lane & 16) ? 8 : 0;
  const int nloc = lane & 15;
  const int which = n0 >> 10;            // 0=q 1=k 2=v (uniform per wave)
  const int nn0 = n0 & (CC - 1);
  const int h = nn0 >> 6;                // uniform per wave
#pragma unroll
  for (int j = 0; j < 4; ++j) {
    const int d = (nn0 & 63) + j * 16 + nloc;
    const float bn = bias[n0 + j * 16 + nloc];
#pragma unroll
    for (int r = 0; r < 8; ++r) {
      const int m = m0 + half + r;
      const int b_ = m / TT, t = m % TT;
      const size_t bh = (size_t)b_ * HH + h;
      const float v = acc[j][r] + bn;
      if (which == 0)      qbuf[(bh * TT + t) * DD + d] = (bf16)(v * 0.125f);
      else if (which == 1) kbuf[(bh * TT + t) * DD + d] = (bf16)v;
      else                 vtbuf[(bh * DD + d) * TT + t] = (bf16)v;
    }
  }
}

// ---------------------------------------------------------------------------
// Flash attention: one wave per (b,h, 16-query tile); online softmax;
// S = Q K^T via WMMA, P V via WMMA (P staged through LDS as bf16 A-fragment).
// Next key-block K/V prefetched (global_prefetch_b8) to overlap HBM/L2
// latency with softmax + WMMA work.
// ---------------------------------------------------------------------------
__global__ void __launch_bounds__(32) attn_kernel(
    const bf16* __restrict__ qbuf,  // [B,H,T,D] pre-scaled
    const bf16* __restrict__ kbuf,  // [B,H,T,D]
    const bf16* __restrict__ vtbuf, // [B,H,D,T]
    bf16* __restrict__ ybuf)        // [B*T, C]
{
  __shared__ bf16 Plds[16][40];     // 16x32 P tile + pad

  const int lane = threadIdx.x & 31;
  const int bh = blockIdx.x;               // 0..B*H-1
  const int q0 = blockIdx.y * 16;
  const int b = bh / HH, h = bh % HH;
  const int half = (lane & 16) ? 8 : 0;
  const int nloc = lane & 15;

  const bf16* Q = qbuf + ((size_t)bh * TT + q0) * DD;
  const bf16* K = kbuf + (size_t)bh * TT * DD;
  const bf16* V = vtbuf + (size_t)bh * DD * TT;

  const v16bf aq0 = load_a_frag(Q, DD, 0);
  const v16bf aq1 = load_a_frag(Q, DD, 32);

  v8f acc[4];
  float mrow[8], lrow[8];
#pragma unroll
  for (int j = 0; j < 4; ++j) acc[j] = (v8f){};
#pragma unroll
  for (int r = 0; r < 8; ++r) { mrow[r] = -1e30f; lrow[r] = 0.0f; }

  const int kend = q0 + 15;
  for (int kb0 = 0; kb0 <= kend; kb0 += 32) {
    // ---- prefetch next key block (one K row / cacheline per lane; V chunks) ----
    if (kb0 + 32 <= kend) {
      __builtin_prefetch(K + (size_t)(kb0 + 32 + lane) * DD, 0, 3);
      __builtin_prefetch(V + (size_t)lane * TT + kb0 + 32, 0, 3);
      __builtin_prefetch(V + (size_t)(lane + 32) * TT + kb0 + 32, 0, 3);
    }
    // ---- S = Q K^T over 32 keys (two 16-key sub-tiles) ----
    v8f s[2];
#pragma unroll
    for (int sub = 0; sub < 2; ++sub) {
      const bf16* Kt = K + (size_t)(kb0 + sub * 16) * DD;  // rows = key positions
      v16bf bk0 = load_bt_frag(Kt, DD, 0);
      v16bf bk1 = load_bt_frag(Kt, DD, 32);
      v8f t = {};
      t = wmma_bf16(aq0, bk0, t);
      t = wmma_bf16(aq1, bk1, t);
      s[sub] = t;
    }
    // ---- causal mask ----
#pragma unroll
    for (int sub = 0; sub < 2; ++sub)
#pragma unroll
      for (int r = 0; r < 8; ++r) {
        const int row = q0 + half + r;
        const int col = kb0 + sub * 16 + nloc;
        s[sub][r] = (col > row) ? -1e30f : s[sub][r];
      }
    // ---- online softmax: new running max per row ----
    float mn[8];
#pragma unroll
    for (int r = 0; r < 8; ++r) mn[r] = fmaxf(mrow[r], fmaxf(s[0][r], s[1][r]));
#pragma unroll
    for (int off = 1; off < 16; off <<= 1)
#pragma unroll
      for (int r = 0; r < 8; ++r) mn[r] = fmaxf(mn[r], __shfl_xor(mn[r], off, 32));
    // rescale accumulators + partial sums
#pragma unroll
    for (int r = 0; r < 8; ++r) {
      const float f = __expf(mrow[r] - mn[r]);
      lrow[r] *= f;
      mrow[r] = mn[r];
#pragma unroll
      for (int j = 0; j < 4; ++j) acc[j][r] *= f;
    }
    // ---- P = exp(S - m), stage to LDS as bf16 ----
#pragma unroll
    for (int sub = 0; sub < 2; ++sub)
#pragma unroll
      for (int r = 0; r < 8; ++r) {
        const float p = __expf(s[sub][r] - mrow[r]);
        lrow[r] += p;
        Plds[half + r][sub * 16 + nloc] = (bf16)p;
      }
    __syncthreads();
    const v16bf ap = load_a_frag(&Plds[0][0], 40, 0);
    // ---- O += P V  (4 tiles over the 64 head dims) ----
#pragma unroll
    for (int j = 0; j < 4; ++j) {
      v16bf bv = load_bt_frag(V + (size_t)(j * 16) * TT, TT, kb0);
      acc[j] = wmma_bf16(ap, bv, acc[j]);
    }
    __syncthreads();
  }

  // ---- finalize: row-sum reduce, divide, store y[B*T, C] as bf16 ----
#pragma unroll
  for (int off = 1; off < 16; off <<= 1)
#pragma unroll
    for (int r = 0; r < 8; ++r) lrow[r] += __shfl_xor(lrow[r], off, 32);
#pragma unroll
  for (int r = 0; r < 8; ++r) {
    const float inv = 1.0f / lrow[r];
    const int t = q0 + half + r;
    const size_t rowbase = ((size_t)b * TT + t) * CC + h * DD;
#pragma unroll
    for (int j = 0; j < 4; ++j)
      ybuf[rowbase + j * 16 + nloc] = (bf16)(acc[j][r] * inv);
  }
}

// ---------------------------------------------------------------------------
// Projection GEMM: [8192 x 1024] @ [1024 x 1024] + bias -> fp32 out
// Same 16x64 wave tile / 32x256 block tile as qkv_gemm_kernel.
// ---------------------------------------------------------------------------
__global__ void __launch_bounds__(256) proj_gemm_kernel(
    const bf16* __restrict__ Yb,    // [B*T, C]
    const bf16* __restrict__ WT,    // [C, C]  (N-major)
    const float* __restrict__ bias, // [C]
    float* __restrict__ out)        // [B*T, C]
{
  const int lane = threadIdx.x & 31;
  const int wave = threadIdx.x >> 5;
  const int m0 = blockIdx.x * 32 + (wave >> 2) * 16;
  const int n0 = blockIdx.y * 256 + (wave & 3) * 64;

  const bf16* Arow = Yb + (size_t)m0 * CC;
  const bf16* Brow = WT + (size_t)n0 * CC;

  v8f acc[4];
#pragma unroll
  for (int j = 0; j < 4; ++j) acc[j] = (v8f){};

  for (int k0 = 0; k0 < CC; k0 += 32) {
    const v16bf a = load_a_frag(Arow, CC, k0);
#pragma unroll
    for (int j = 0; j < 4; ++j) {
      v16bf b = load_bt_frag(Brow + (size_t)(j * 16) * CC, CC, k0);
      acc[j] = wmma_bf16(a, b, acc[j]);
    }
  }

  const int half = (lane & 16) ? 8 : 0;
#pragma unroll
  for (int j = 0; j < 4; ++j) {
    const int n = n0 + j * 16 + (lane & 15);
    const float bn = bias[n];
#pragma unroll
    for (int r = 0; r < 8; ++r) {
      const int m = m0 + half + r;
      out[(size_t)m * CC + n] = acc[j][r] + bn;
    }
  }
}

// ---------------------------------------------------------------------------
// Launch
// ---------------------------------------------------------------------------
extern "C" void kernel_launch(void* const* d_in, const int* in_sizes, int n_in,
                              void* d_out, int out_size, void* d_ws, size_t ws_size,
                              hipStream_t stream) {
  const float* x      = (const float*)d_in[0];
  const float* W_qkv  = (const float*)d_in[1];
  const float* b_qkv  = (const float*)d_in[2];
  const float* W_proj = (const float*)d_in[3];
  const float* b_proj = (const float*)d_in[4];
  float* out = (float*)d_out;

  char* ws = (char*)d_ws;
  size_t off = 0;
  auto alloc = [&](size_t bytes) -> char* {
    char* p = ws + off;
    off += (bytes + 255) & ~(size_t)255;
    return p;
  };

  const size_t nBTC = (size_t)BB * TT * CC;  // 8,388,608
  bf16* xb     = (bf16*)alloc(nBTC * sizeof(bf16));
  bf16* wqkvT  = (bf16*)alloc((size_t)3 * CC * CC * sizeof(bf16));
  bf16* wprojT = (bf16*)alloc((size_t)CC * CC * sizeof(bf16));
  bf16* qbuf   = (bf16*)alloc(nBTC * sizeof(bf16));
  bf16* kbuf   = (bf16*)alloc(nBTC * sizeof(bf16));
  bf16* vtbuf  = (bf16*)alloc(nBTC * sizeof(bf16));
  bf16* ybuf   = (bf16*)alloc(nBTC * sizeof(bf16));
  (void)ws_size; (void)in_sizes; (void)n_in; (void)out_size;

  // 1) conversions
  cvt_bf16_kernel<<<2048, 256, 0, stream>>>(x, xb, (long)nBTC);
  transpose_cvt_kernel<<<2048, 256, 0, stream>>>(W_qkv, wqkvT, CC, 3 * CC);
  transpose_cvt_kernel<<<1024, 256, 0, stream>>>(W_proj, wprojT, CC, CC);

  // 2) QKV GEMM: M = B*T = 8192, N = 3C = 3072 (block tile 32x256)
  qkv_gemm_kernel<<<dim3(8192 / 32, 3072 / 256), 256, 0, stream>>>(
      xb, wqkvT, b_qkv, qbuf, kbuf, vtbuf);

  // 3) attention: one wave per (b,h,16-query tile)
  attn_kernel<<<dim3(BB * HH, TT / 16), 32, 0, stream>>>(qbuf, kbuf, vtbuf, ybuf);

  // 4) projection GEMM: M = 8192, N = 1024 (block tile 32x256)
  proj_gemm_kernel<<<dim3(8192 / 32, 1024 / 256), 256, 0, stream>>>(
      ybuf, wprojT, b_proj, out);
}
